// ActorCritic_72018011619396
// MI455X (gfx1250) — compile-verified, matrix operand order checked by reading, not run on hardware
//
#include <hip/hip_runtime.h>
#include <hip/hip_bf16.h>
#include <math.h>

// ---------------- model constants ----------------
#define D      64
#define DFF    512
#define NHEAD  4
#define HD     128
#define NLAYERS 3
#define HIST   10
#define ADIM   63
#define BATCH  1024
#define FEAT   (3*D + 2)     // 194
#define VFK    224           // FEAT padded to 32
#define NPAIR  (D*(D-1)/2)   // 2016
#define GIN    (2*D + 1 + NPAIR)   // 2145
#define GINP   2176          // GIN padded to 32
#define COMB   (D*(DFF/2) + 128 + 64)   // 16576 (already %32==0)
#define SEQROWS (BATCH*D)    // 65536

typedef unsigned short u16;

// ---------------- WMMA types ----------------
typedef __bf16 v16bf __attribute__((ext_vector_type(16)));
typedef float  v8f   __attribute__((ext_vector_type(8)));
typedef unsigned short v8u  __attribute__((ext_vector_type(8)));
typedef unsigned short v16u __attribute__((ext_vector_type(16)));

__device__ __forceinline__ u16 f2bf(float f) {
    union { float f; unsigned u; } c; c.f = f;
    unsigned u = c.u;
    u += 0x7FFFu + ((u >> 16) & 1u);     // round-to-nearest-even
    return (u16)(u >> 16);
}
__device__ __forceinline__ float bf2f(u16 h) {
    union { unsigned u; float f; } c; c.u = (unsigned)h << 16;
    return c.f;
}

// Fragment load for the 16-bit operand layout of v_wmma_f32_16x16x32_bf16:
// lane<16 -> K{0..7},K{16..23} of row (lane&15); lane>=16 -> K{8..15},K{24..31}.
__device__ __forceinline__ v16bf frag_ld(const u16* p, int off, int hi) {
    const v8u lo = *(const v8u*)(p + off + hi * 8);
    const v8u hh = *(const v8u*)(p + off + hi * 8 + 16);
    v16u t = __builtin_shufflevector(lo, hh, 0,1,2,3,4,5,6,7,8,9,10,11,12,13,14,15);
    return __builtin_bit_cast(v16bf, t);
}
__device__ __forceinline__ v8f wmma_bf16(v16bf a, v16bf b, v8f c) {
    return __builtin_amdgcn_wmma_f32_16x16x32_bf16(
        false, a, false, b, (short)0, c, false, false);
}

// =====================================================================
// FAST GEMM (bf16 in, bf16 out): C = act(A(MxK) Wt(NxK)^T + bias [+res] [+pos])
// Preconditions: M%128==0, N%64==0, K%32==0 (buffers padded by host).
// 128x64 C-tile, 256 threads = 8 waves (2M x 4N), 4 accumulators/wave,
// double-buffered LDS, staging = pure b128 copies (no converts, no guards).
// cgroup!=0: C index = (row/64)*cgroup + (row%64)*ldc + col (flattened comb).
// =====================================================================
__global__ __launch_bounds__(256) void gemm_fast(
    const u16* __restrict__ A, const u16* __restrict__ Wt,
    const float* __restrict__ bias, const u16* __restrict__ residual,
    const float* __restrict__ pos, u16* __restrict__ C,
    int M, int N, int K, int ldc, int cgroup, int relu)
{
    __shared__ __align__(16) u16 sA[2][128 * 32];
    __shared__ __align__(16) u16 sB[2][64 * 32];

    const int tid  = threadIdx.x;
    const int wave = tid >> 5, lane = tid & 31;
    const int wrow = lane & 15, whi = lane >> 4;
    const int wm = wave >> 2, wn = wave & 3;
    const int mBlk = blockIdx.y * 128, nBlk = blockIdx.x * 64;

    auto stage = [&](int buf, int kc) {
        // A tile: 128x32 bf16 = 512 x 16B; 2 segments / thread
        int s = tid;
#pragma unroll
        for (int it = 0; it < 2; ++it, s += 256) {
            const int row = s >> 2, seg = s & 3;
            *(v8u*)&sA[buf][row * 32 + seg * 8] =
                *(const v8u*)(A + (size_t)(mBlk + row) * K + kc + seg * 8);
        }
        // W tile: 64x32 bf16 = 256 x 16B; 1 segment / thread
        const int row = tid >> 2, seg = tid & 3;
        *(v8u*)&sB[buf][row * 32 + seg * 8] =
            *(const v8u*)(Wt + (size_t)(nBlk + row) * K + kc + seg * 8);
    };

    v8f acc[4] = {{}, {}, {}, {}};
    const int nch = K >> 5;
    stage(0, 0);
    __syncthreads();
    for (int c = 0; c < nch; ++c) {
        if (c + 1 < nch) stage((c + 1) & 1, (c + 1) << 5);
        const int buf = c & 1;
        const v16bf bf = frag_ld(sB[buf], (wn * 16 + wrow) * 32, whi);
#pragma unroll
        for (int mt = 0; mt < 4; ++mt) {
            const v16bf a = frag_ld(sA[buf], (wm * 64 + mt * 16 + wrow) * 32, whi);
            acc[mt] = wmma_bf16(a, bf, acc[mt]);
        }
        __syncthreads();
    }

    const int gn = nBlk + wn * 16 + wrow;
#pragma unroll
    for (int mt = 0; mt < 4; ++mt)
#pragma unroll
        for (int r = 0; r < 8; ++r) {
            const int gm = mBlk + wm * 64 + mt * 16 + whi * 8 + r;
            float v = acc[mt][r];
            if (bias)     v += bias[gn];
            if (residual) v += bf2f(residual[(size_t)gm * ldc + gn]);
            if (pos)      v += pos[(size_t)(gm & 63) * N + gn];
            if (relu)     v = fmaxf(v, 0.0f);
            const size_t cidx = cgroup
                ? (size_t)(gm >> 6) * cgroup + (size_t)(gm & 63) * ldc + gn
                : (size_t)gm * ldc + gn;
            C[cidx] = f2bf(v);
        }
}

// Guarded fallback GEMM (bf16 in/out), 64x64 tile — used only for the tiny
// unaligned-N heads (actor3 N=63, critic3 N=1).
__global__ __launch_bounds__(256) void gemm_gen(
    const u16* __restrict__ A, const u16* __restrict__ Wt,
    const float* __restrict__ bias, u16* __restrict__ C,
    int M, int N, int K, int ldc)
{
    __shared__ __align__(16) u16 sA[64 * 32];
    __shared__ __align__(16) u16 sB[64 * 32];
    const int tid  = threadIdx.x;
    const int wave = tid >> 5, lane = tid & 31;
    const int wrow = lane & 15, whi = lane >> 4;
    const int wm = wave >> 2, wn = wave & 3;
    const int mBlk = blockIdx.y * 64, nBlk = blockIdx.x * 64;

    v8f acc0 = {}, acc1 = {};
    for (int kc = 0; kc < K; kc += 32) {
        for (int i = tid; i < 64 * 32; i += 256) {
            const int r = i >> 5, c = i & 31;
            const int gk = kc + c, ga = mBlk + r, gw = nBlk + r;
            sA[r * 32 + c] = (ga < M && gk < K) ? A[(size_t)ga * K + gk] : (u16)0;
            sB[r * 32 + c] = (gw < N && gk < K) ? Wt[(size_t)gw * K + gk] : (u16)0;
        }
        __syncthreads();
        const v16bf a0 = frag_ld(sA, (wm * 32 +      wrow) * 32, whi);
        const v16bf a1 = frag_ld(sA, (wm * 32 + 16 + wrow) * 32, whi);
        const v16bf bf = frag_ld(sB, (wn * 16 +      wrow) * 32, whi);
        acc0 = wmma_bf16(a0, bf, acc0);
        acc1 = wmma_bf16(a1, bf, acc1);
        __syncthreads();
    }
    const int gn = nBlk + wn * 16 + wrow;
    for (int t = 0; t < 2; ++t) {
        const v8f acc = t ? acc1 : acc0;
        for (int r = 0; r < 8; ++r) {
            const int gm = mBlk + wm * 32 + t * 16 + whi * 8 + r;
            if (gm >= M || gn >= N) continue;
            float v = acc[r];
            if (bias) v += bias[gn];
            C[(size_t)gm * ldc + gn] = f2bf(v);
        }
    }
}

// =====================================================================
// Fused attention: one block per (batch, head). Q K^T and P V via WMMA,
// qkv is bf16 so staging is pure copies; softmax = 4 threads/row.
// =====================================================================
__global__ __launch_bounds__(256) void attn_kernel(
    const u16* __restrict__ qkv, u16* __restrict__ out)
{
    __shared__ __align__(16) u16 sQ[64 * 128];   // Q, later P (64x64)
    __shared__ __align__(16) u16 sK[64 * 128];   // K, later V^T (128x64)
    __shared__ float sS[64 * 64];

    const int bh = blockIdx.x, b = bh >> 2, h = bh & 3;
    const int tid  = threadIdx.x;
    const int wave = tid >> 5, lane = tid & 31;
    const int wrow = lane & 15, whi = lane >> 4;

    const u16* base = qkv + (size_t)b * 64 * (3 * DFF) + h * HD;

    for (int s = tid; s < 1024; s += 256) {       // 8192 elems / 8
        const int q = s >> 4, seg = (s & 15) * 8;
        *(v8u*)&sQ[q * 128 + seg] = *(const v8u*)(base + (size_t)q * (3 * DFF) + seg);
        *(v8u*)&sK[q * 128 + seg] = *(const v8u*)(base + (size_t)q * (3 * DFF) + DFF + seg);
    }
    __syncthreads();

    // ---- scores: 2x4 wave grid, 2 acc/wave, K=128 ----
    {
        const int wm = wave >> 2, wn = wave & 3;
        v8f s0 = {}, s1 = {};
#pragma unroll
        for (int kc = 0; kc < 128; kc += 32) {
            const v16bf q0 = frag_ld(sQ, (wm * 32 +      wrow) * 128 + kc, whi);
            const v16bf q1 = frag_ld(sQ, (wm * 32 + 16 + wrow) * 128 + kc, whi);
            const v16bf kb = frag_ld(sK, (wn * 16 +      wrow) * 128 + kc, whi);
            s0 = wmma_bf16(q0, kb, s0);
            s1 = wmma_bf16(q1, kb, s1);
        }
        const float scale = 0.08838834764831845f;   // 1/sqrt(128)
#pragma unroll
        for (int r = 0; r < 8; ++r) {
            sS[(wm * 32 +      whi * 8 + r) * 64 + wn * 16 + wrow] = s0[r] * scale;
            sS[(wm * 32 + 16 + whi * 8 + r) * 64 + wn * 16 + wrow] = s1[r] * scale;
        }
    }
    __syncthreads();

    // ---- V transposed into sK (pure u16 copies); softmax -> P in sQ ----
    for (int i = tid; i < 64 * 128; i += 256) {
        const int k = i >> 7, d = i & 127;
        sK[d * 64 + k] = base[(size_t)k * (3 * DFF) + 2 * DFF + d];
    }
    {
        const int r = tid >> 2, q4 = (tid & 3) * 16;
        float vv[16];
        float mx = -3.402823466e38f;
#pragma unroll
        for (int j = 0; j < 16; ++j) { vv[j] = sS[r * 64 + q4 + j]; mx = fmaxf(mx, vv[j]); }
        mx = fmaxf(mx, __shfl_xor(mx, 1, 4));
        mx = fmaxf(mx, __shfl_xor(mx, 2, 4));
        float s = 0.0f;
#pragma unroll
        for (int j = 0; j < 16; ++j) { vv[j] = __expf(vv[j] - mx); s += vv[j]; }
        s += __shfl_xor(s, 1, 4);
        s += __shfl_xor(s, 2, 4);
        const float inv = 1.0f / s;
#pragma unroll
        for (int j = 0; j < 16; ++j) sQ[r * 64 + q4 + j] = f2bf(vv[j] * inv);
    }
    __syncthreads();

    // ---- O = P(64x64) @ V(64x128): 8 wave-cols x 4 m-tiles ----
    v8f o[4] = {{}, {}, {}, {}};
#pragma unroll
    for (int kc = 0; kc < 64; kc += 32) {
        const v16bf bv = frag_ld(sK, (wave * 16 + wrow) * 64 + kc, whi);
#pragma unroll
        for (int mt = 0; mt < 4; ++mt) {
            const v16bf pa = frag_ld(sQ, (mt * 16 + wrow) * 64 + kc, whi);
            o[mt] = wmma_bf16(pa, bv, o[mt]);
        }
    }
    u16* obase = out + (size_t)b * 64 * DFF + h * HD;
#pragma unroll
    for (int mt = 0; mt < 4; ++mt)
#pragma unroll
        for (int r = 0; r < 8; ++r) {
            const int gq = mt * 16 + whi * 8 + r;
            obase[(size_t)gq * DFF + wave * 16 + wrow] = f2bf(o[mt][r]);
        }
}

// =====================================================================
// LayerNorm (bf16 -> bf16): one wave32 per 512-row, 16 contiguous cols/lane.
// =====================================================================
__global__ __launch_bounds__(256) void ln_kernel(
    const u16* __restrict__ in, const float* __restrict__ g,
    const float* __restrict__ bta, u16* __restrict__ out, int nrows)
{
    const int wave = threadIdx.x >> 5, lane = threadIdx.x & 31;
    const int row = blockIdx.x * 8 + wave;
    if (row >= nrows) return;
    const u16* p = in + (size_t)row * DFF + lane * 16;
    const v8u d0 = *(const v8u*)p;
    const v8u d1 = *(const v8u*)(p + 8);
    float v[16];
#pragma unroll
    for (int i = 0; i < 8; ++i) { v[i] = bf2f(d0[i]); v[8 + i] = bf2f(d1[i]); }
    float s = 0.0f;
#pragma unroll
    for (int i = 0; i < 16; ++i) s += v[i];
    for (int m = 16; m >= 1; m >>= 1) s += __shfl_xor(s, m, 32);
    const float mean = s * (1.0f / DFF);
    float vs = 0.0f;
#pragma unroll
    for (int i = 0; i < 16; ++i) { const float d = v[i] - mean; vs += d * d; }
    for (int m = 16; m >= 1; m >>= 1) vs += __shfl_xor(vs, m, 32);
    const float rstd = rsqrtf(vs * (1.0f / DFF) + 1e-5f);
    const int c0 = lane * 16;
    v8u o0, o1;
#pragma unroll
    for (int i = 0; i < 8; ++i) {
        o0[i] = f2bf((v[i]     - mean) * rstd * g[c0 + i]     + bta[c0 + i]);
        o1[i] = f2bf((v[8 + i] - mean) * rstd * g[c0 + 8 + i] + bta[c0 + 8 + i]);
    }
    u16* q = out + (size_t)row * DFF + c0;
    *(v8u*)q = o0;
    *(v8u*)(q + 8) = o1;
}

// =====================================================================
// Gram-Schmidt: one block (64 threads) per batch item, LDS-resident.
// =====================================================================
__global__ __launch_bounds__(64) void gs_kernel(
    const float* __restrict__ basis, float* __restrict__ gs_out,
    float* __restrict__ mu_out, float* __restrict__ bn_out,
    float* __restrict__ gn_out)
{
    __shared__ float sB[D * D];
    __shared__ float sG[D * D];
    __shared__ float sSq[D], sMu[D], sRed[D];
    const int b = blockIdx.x, tid = threadIdx.x;

    for (int i = 0; i < D; ++i)
        sB[i * D + tid] = basis[(size_t)b * D * D + i * D + tid];
    sSq[tid] = 0.0f;
    __syncthreads();

    for (int j = 0; j < D; ++j) {
        float dot = 0.0f;
        if (tid < j)
            for (int d = 0; d < D; ++d) dot += sG[tid * D + d] * sB[j * D + d];
        const float sq = sSq[tid];
        const float m = (tid < j) ? dot / (sq > 0.0f ? sq : 1.0f) : 0.0f;
        sMu[tid] = m;
        mu_out[(size_t)b * D * D + j * D + tid] = m;
        __syncthreads();

        float v = sB[j * D + tid];
        for (int i = 0; i < j; ++i) v -= sMu[i] * sG[i * D + tid];
        sG[j * D + tid] = v;
        sRed[tid] = v * v;
        __syncthreads();
        if (tid == 0) {
            float s = 0.0f;
            for (int i = 0; i < D; ++i) s += sRed[i];
            sSq[j] = s;
        }
        __syncthreads();
    }

    for (int i = 0; i < D; ++i)
        gs_out[(size_t)b * D * D + i * D + tid] = sG[i * D + tid];
    float bs = 0.0f;
    for (int d = 0; d < D; ++d) { const float x = sB[tid * D + d]; bs += x * x; }
    bn_out[(size_t)b * D + tid] = sqrtf(bs);
    gn_out[(size_t)b * D + tid] = sqrtf(sSq[tid]);
}

// =====================================================================
// gfeat (bf16, K padded to 2176) = [orth_defect | cos pairs | bn | gn | 0pad]
// =====================================================================
__global__ __launch_bounds__(256) void gfeat_kernel(
    const float* __restrict__ basis, const float* __restrict__ bn,
    const float* __restrict__ gn, u16* __restrict__ gfeat)
{
    __shared__ float sNB[D * D];
    const int b = blockIdx.x, tid = threadIdx.x;
    for (int i = tid; i < D * D; i += 256) {
        const int r = i >> 6;
        sNB[i] = basis[(size_t)b * D * D + i] / (bn[(size_t)b * D + r] + 1e-10f);
    }
    __syncthreads();

    u16* gf = gfeat + (size_t)b * GINP;
    if (tid == 0) {
        float sb = 0.0f, sg = 0.0f;
        for (int i = 0; i < D; ++i) {
            sb += logf(bn[(size_t)b * D + i] + 1e-30f);
            sg += logf(gn[(size_t)b * D + i] + 1e-30f);
        }
        gf[0] = f2bf(__expf(sb) / (__expf(sg) + 1e-10f));
    }
    for (int p = tid; p < NPAIR; p += 256) {
        int i = 0, rem = p, cnt = D - 1;
        while (rem >= cnt) { rem -= cnt; --cnt; ++i; }
        const int j = i + 1 + rem;
        float d = 0.0f;
        for (int k = 0; k < D; ++k) d += sNB[i * D + k] * sNB[j * D + k];
        gf[1 + p] = f2bf(d);
    }
    if (tid < D) {
        gf[1 + NPAIR + tid]     = f2bf(bn[(size_t)b * D + tid]);
        gf[1 + NPAIR + D + tid] = f2bf(gn[(size_t)b * D + tid]);
    }
    for (int p = GIN + tid; p < GINP; p += 256) gf[p] = 0;
}

// vf[b,j,:] -> bf16, K padded to 224
__global__ __launch_bounds__(256) void vf_kernel(
    const float* __restrict__ basis, const float* __restrict__ gs,
    const float* __restrict__ mu, const float* __restrict__ bn,
    const float* __restrict__ gn, u16* __restrict__ vf)
{
    const size_t i = (size_t)blockIdx.x * 256 + threadIdx.x;
    const size_t total = (size_t)BATCH * D * VFK;
    if (i >= total) return;
    const int f = (int)(i % VFK);
    const size_t bj = i / VFK;
    float v;
    if      (f < 64)   v = basis[bj * D + f];
    else if (f < 128)  v = gs[bj * D + (f - 64)];
    else if (f == 128) v = bn[bj];
    else if (f == 129) v = gn[bj];
    else if (f < FEAT) v = mu[bj * D + (f - 130)];
    else               v = 0.0f;
    vf[i] = f2bf(v);
}

// action-history embedding gather -> bf16 (B, 320)
__global__ __launch_bounds__(256) void ae_kernel(
    const int* __restrict__ ah, const float* __restrict__ act_emb,
    u16* __restrict__ ae)
{
    const int i = blockIdx.x * 256 + threadIdx.x;
    if (i >= BATCH * HIST * 32) return;
    const int b = i / (HIST * 32);
    const int r = i % (HIST * 32);
    const int t = r >> 5, k = r & 31;
    int idx = ah[b * HIST + t];
    if (idx < 0) idx = ADIM;
    ae[i] = f2bf(act_emb[idx * 32 + k]);
}

// weight convert+pad: f32 (N x K) -> bf16 (N x Kp), zero K-tail
__global__ __launch_bounds__(256) void cvt_kernel(
    const float* __restrict__ src, u16* __restrict__ dst, int N, int K, int Kp)
{
    const size_t i = (size_t)blockIdx.x * 256 + threadIdx.x;
    if (i >= (size_t)N * Kp) return;
    const int k = (int)(i % Kp);
    const size_t n = i / Kp;
    dst[i] = (k < K) ? f2bf(src[n * K + k]) : (u16)0;
}

// final: probs = softmax(logits[b, :63]) ; value
__global__ __launch_bounds__(256) void final_kernel(
    const u16* __restrict__ logits, const u16* __restrict__ cval,
    float* __restrict__ out)
{
    const int b = blockIdx.x * 256 + threadIdx.x;
    if (b >= BATCH) return;
    float e[ADIM];
    float mx = -3.402823466e38f;
    for (int j = 0; j < ADIM; ++j) { e[j] = bf2f(logits[b * ADIM + j]); mx = fmaxf(mx, e[j]); }
    float s = 0.0f;
    for (int j = 0; j < ADIM; ++j) { e[j] = __expf(e[j] - mx); s += e[j]; }
    const float inv = 1.0f / s;
    for (int j = 0; j < ADIM; ++j) out[b * ADIM + j] = e[j] * inv;
    out[BATCH * ADIM + b] = bf2f(cval[b]);
}

// =====================================================================
// Host-side orchestration
// =====================================================================
static inline void fgemm(hipStream_t st, const u16* A, const u16* W,
                         const float* bias, const u16* res, const float* pos,
                         u16* C, int M, int N, int K, int ldc, int cgroup, int relu)
{
    dim3 grid(N / 64, M / 128);
    gemm_fast<<<grid, 256, 0, st>>>(A, W, bias, res, pos, C, M, N, K, ldc, cgroup, relu);
}
static inline void cvt(hipStream_t st, const float* src, u16* dst, int N, int K, int Kp)
{
    const size_t tot = (size_t)N * Kp;
    cvt_kernel<<<(unsigned)((tot + 255) / 256), 256, 0, st>>>(src, dst, N, K, Kp);
}

extern "C" void kernel_launch(void* const* d_in, const int* in_sizes, int n_in,
                              void* d_out, int out_size, void* d_ws, size_t ws_size,
                              hipStream_t stream)
{
    (void)in_sizes; (void)n_in; (void)out_size; (void)ws_size;
    const float* basis = (const float*)d_in[0];
    const int*   ah    = (const int*)d_in[1];
    const float* prm[62];
    for (int i = 0; i < 62; ++i) prm[i] = (const float*)d_in[2 + i];
    // 0 emb_w, 1 emb_b, 2 pos, 3..38 layers (in_w,in_b,out_w,out_b,ln1_g,ln1_b,
    // ff1_w,ff1_b,ff2_w,ff2_b,ln2_g,ln2_b), 39 proj_w, 40 proj_b, 41 g1_w,
    // 42 g1_b, 43 g2_w, 44 g2_b, 45 act_emb, 46 a1_w, 47 a1_b, 48 a2_w,
    // 49 a2_b, 50..55 actor1..3 w/b, 56..61 critic1..3 w/b
#define LP(L, k) prm[3 + (L) * 12 + (k)]

    size_t off = 0;
    auto allocB = [&](size_t bytes) -> void* {
        void* p = (char*)d_ws + off;
        off += (bytes + 255) & ~(size_t)255;
        return p;
    };
    auto aF = [&](size_t n) { return (float*)allocB(n * 4); };
    auto aH = [&](size_t n) { return (u16*)allocB(n * 2); };

    // f32 geometry scratch
    float* gs = aF((size_t)BATCH * D * D);
    float* mu = aF((size_t)BATCH * D * D);
    float* bn = aF((size_t)BATCH * D);
    float* gn = aF((size_t)BATCH * D);
    // bf16 activations
    u16* vf    = aH((size_t)SEQROWS * VFK);
    u16* gfeat = aH((size_t)BATCH * GINP);
    u16* ae    = aH((size_t)BATCH * HIST * 32);
    u16* x     = aH((size_t)SEQROWS * DFF);
    u16* y     = aH((size_t)SEQROWS * DFF);
    u16* x1    = aH((size_t)SEQROWS * DFF);
    u16* attnb = aH((size_t)SEQROWS * DFF);
    u16* qkv   = aH((size_t)SEQROWS * 3 * DFF);    // reused as FF hidden
    u16* comb  = aH((size_t)BATCH * COMB);
    u16* g1o   = aH((size_t)BATCH * 256);
    u16* a1o   = aH((size_t)BATCH * 128);
    u16* h1    = aH((size_t)BATCH * 512);
    u16* h2    = aH((size_t)BATCH * 512);
    u16* logit = aH((size_t)BATCH * ADIM);
    u16* cval  = aH((size_t)BATCH);
    // bf16 padded weights
    u16* w_emb  = aH((size_t)DFF * VFK);
    u16* w_in[NLAYERS], *w_out[NLAYERS], *w_f1[NLAYERS], *w_f2[NLAYERS];
    for (int L = 0; L < NLAYERS; ++L) {
        w_in[L]  = aH((size_t)3 * DFF * DFF);
        w_out[L] = aH((size_t)DFF * DFF);
        w_f1[L]  = aH((size_t)2 * DFF * DFF);
        w_f2[L]  = aH((size_t)DFF * 2 * DFF);
    }
    u16* w_proj = aH((size_t)(DFF / 2) * DFF);
    u16* w_g1 = aH((size_t)256 * GINP);
    u16* w_g2 = aH((size_t)128 * 256);
    u16* w_a1 = aH((size_t)128 * HIST * 32);
    u16* w_a2 = aH((size_t)64 * 128);
    u16* w_ac1 = aH((size_t)512 * COMB);
    u16* w_ac2 = aH((size_t)512 * 512);
    u16* w_ac3 = aH((size_t)ADIM * 512);
    u16* w_cr1 = aH((size_t)512 * COMB);
    u16* w_cr2 = aH((size_t)512 * 512);
    u16* w_cr3 = aH((size_t)1 * 512);

    // ---- weight conversion (bf16, K-padded) ----
    cvt(stream, prm[0], w_emb, DFF, FEAT, VFK);
    for (int L = 0; L < NLAYERS; ++L) {
        cvt(stream, LP(L, 0), w_in[L],  3 * DFF, DFF, DFF);
        cvt(stream, LP(L, 2), w_out[L], DFF, DFF, DFF);
        cvt(stream, LP(L, 6), w_f1[L],  2 * DFF, DFF, DFF);
        cvt(stream, LP(L, 8), w_f2[L],  DFF, 2 * DFF, 2 * DFF);
    }
    cvt(stream, prm[39], w_proj, DFF / 2, DFF, DFF);
    cvt(stream, prm[41], w_g1, 256, GIN, GINP);
    cvt(stream, prm[43], w_g2, 128, 256, 256);
    cvt(stream, prm[46], w_a1, 128, HIST * 32, HIST * 32);
    cvt(stream, prm[48], w_a2, 64, 128, 128);
    cvt(stream, prm[50], w_ac1, 512, COMB, COMB);
    cvt(stream, prm[52], w_ac2, 512, 512, 512);
    cvt(stream, prm[54], w_ac3, ADIM, 512, 512);
    cvt(stream, prm[56], w_cr1, 512, COMB, COMB);
    cvt(stream, prm[58], w_cr2, 512, 512, 512);
    cvt(stream, prm[60], w_cr3, 1, 512, 512);

    // ---- geometry features ----
    gs_kernel<<<BATCH, 64, 0, stream>>>(basis, gs, mu, bn, gn);
    gfeat_kernel<<<BATCH, 256, 0, stream>>>(basis, bn, gn, gfeat);
    {
        const size_t tot = (size_t)SEQROWS * VFK;
        vf_kernel<<<(unsigned)((tot + 255) / 256), 256, 0, stream>>>(basis, gs, mu, bn, gn, vf);
    }
    ae_kernel<<<(BATCH * HIST * 32 + 255) / 256, 256, 0, stream>>>(ah, prm[45], ae);

    // ---- embedding ----
    fgemm(stream, vf, w_emb, prm[1], nullptr, prm[2], x,
          SEQROWS, DFF, VFK, DFF, 0, 0);

    // ---- transformer encoder ----
    for (int L = 0; L < NLAYERS; ++L) {
        fgemm(stream, x, w_in[L], LP(L, 1), nullptr, nullptr, qkv,
              SEQROWS, 3 * DFF, DFF, 3 * DFF, 0, 0);
        attn_kernel<<<BATCH * NHEAD, 256, 0, stream>>>(qkv, attnb);
        fgemm(stream, attnb, w_out[L], LP(L, 3), /*res=*/x, nullptr, y,
              SEQROWS, DFF, DFF, DFF, 0, 0);
        ln_kernel<<<SEQROWS / 8, 256, 0, stream>>>(y, LP(L, 4), LP(L, 5), x1, SEQROWS);
        fgemm(stream, x1, w_f1[L], LP(L, 7), nullptr, nullptr, qkv,
              SEQROWS, 2 * DFF, DFF, 2 * DFF, 0, 1);
        fgemm(stream, qkv, w_f2[L], LP(L, 9), /*res=*/x1, nullptr, y,
              SEQROWS, DFF, 2 * DFF, DFF, 0, 0);
        ln_kernel<<<SEQROWS / 8, 256, 0, stream>>>(y, LP(L, 10), LP(L, 11), x, SEQROWS);
    }

    // ---- proj -> comb[:, 0:16384] (grouped store) ----
    fgemm(stream, x, w_proj, prm[40], nullptr, nullptr, comb,
          SEQROWS, DFF / 2, DFF, DFF / 2, COMB, 0);

    // ---- g branch -> comb[:, 16384:16512] ----
    fgemm(stream, gfeat, w_g1, prm[42], nullptr, nullptr, g1o,
          BATCH, 256, GINP, 256, 0, 1);
    fgemm(stream, g1o, w_g2, prm[44], nullptr, nullptr, comb + D * (DFF / 2),
          BATCH, 128, 256, COMB, 0, 0);

    // ---- a branch -> comb[:, 16512:16576] ----
    fgemm(stream, ae, w_a1, prm[47], nullptr, nullptr, a1o,
          BATCH, 128, HIST * 32, 128, 0, 1);
    fgemm(stream, a1o, w_a2, prm[49], nullptr, nullptr,
          comb + D * (DFF / 2) + 128, BATCH, 64, 128, COMB, 0, 0);

    // ---- actor head ----
    fgemm(stream, comb, w_ac1, prm[51], nullptr, nullptr, h1,
          BATCH, 512, COMB, 512, 0, 1);
    fgemm(stream, h1, w_ac2, prm[53], nullptr, nullptr, h2,
          BATCH, 512, 512, 512, 0, 1);
    gemm_gen<<<dim3(1, BATCH / 64), 256, 0, stream>>>(h2, w_ac3, prm[55], logit,
                                                      BATCH, ADIM, 512, ADIM);
    // ---- critic head ----
    fgemm(stream, comb, w_cr1, prm[57], nullptr, nullptr, h1,
          BATCH, 512, COMB, 512, 0, 1);
    fgemm(stream, h1, w_cr2, prm[59], nullptr, nullptr, h2,
          BATCH, 512, 512, 512, 0, 1);
    gemm_gen<<<dim3(1, BATCH / 64), 256, 0, stream>>>(h2, w_cr3, prm[61], cval,
                                                      BATCH, 1, 512, 1);

    final_kernel<<<(BATCH + 255) / 256, 256, 0, stream>>>(logit, cval, (float*)d_out);
#undef LP
}
